// MAMBA_82540681495076
// MI455X (gfx1250) — matrix operand order repaired and checked by reading, not compiled, not run on hardware
//
#include <hip/hip_runtime.h>
#include <hip/hip_bf16.h>
#include <math.h>

// ---------------- problem constants ----------------
constexpr int cB   = 4;
constexpr int cDM  = 256;
constexpr int cL   = 2048;
constexpr int cN   = 16;    // SSM state dim
constexpr int cK   = 4;     // conv kernel
constexpr int cDIN = 512;   // EXP * DM
constexpr int cDTR = 16;    // (DM+15)/16
constexpr int cXZC = 2 * cDIN;          // 1024 columns of xz
constexpr int cXDC = cDTR + 2 * cN;     // 48 columns of x_dbl
constexpr int cROWS = cB * cL;          // 8192 GEMM rows

typedef __attribute__((ext_vector_type(16))) _Float16 v16h;
typedef __attribute__((ext_vector_type(8)))  _Float16 v8h;
typedef __attribute__((ext_vector_type(8)))  float    v8f;

// ---------------- f16 fragment loaders: 2x b128 each, zero in-loop conversions ----------------
__device__ __forceinline__ v16h combine8(v8h lo, v8h hi) {
  v16h f;
#pragma unroll
  for (int i = 0; i < 8; ++i) { f[i] = lo[i]; f[i + 8] = hi[i]; }
  return f;
}

// A 16x32 f16: lane<16 -> M=lane, K=k0+{0..15}; lane>=16 -> M=lane-16, K=k0+{16..31}
__device__ __forceinline__ v16h frag_a16(const _Float16* __restrict__ prow, int k0, int half) {
  const _Float16* p = prow + k0 + half * 16;
  return combine8(*(const v8h*)p, *(const v8h*)(p + 8));
}

// B 32x16 f16: column N = n0+(lane&15); half 0 -> K=k0..k0+15, half 1 -> K=k0+16..k0+31; B(k,n)=W[n*ldw+k]
__device__ __forceinline__ v16h frag_b16(const _Float16* __restrict__ prow, int k0, int half) {
  const _Float16* p = prow + k0 + half * 16;
  return combine8(*(const v8h*)p, *(const v8h*)(p + 8));
}

// Tail fragments for KMAX % 32 == 16 (delta projection, KMAX=16):
// A and B each need one per-half runtime branch.
template <int KMAX>
__device__ __forceinline__ v16h frag_a16_tail(const _Float16* __restrict__ prow, int half) {
  constexpr int k0 = (KMAX / 32) * 32;
  v16h f;
#pragma unroll
  for (int i = 0; i < 16; ++i) f[i] = (_Float16)0.f;
  if (k0 + half * 16 + 16 <= KMAX) {
    const _Float16* p = prow + k0 + half * 16;
    f = combine8(*(const v8h*)p, *(const v8h*)(p + 8));
  }
  return f;
}

template <int KMAX>
__device__ __forceinline__ v16h frag_b16_tail(const _Float16* __restrict__ prow, int half) {
  constexpr int k0 = (KMAX / 32) * 32;
  v16h f;
#pragma unroll
  for (int i = 0; i < 16; ++i) f[i] = (_Float16)0.f;
  if (k0 + half * 16 + 16 <= KMAX) {
    const _Float16* p = prow + k0 + half * 16;
    f = combine8(*(const v8h*)p, *(const v8h*)(p + 8));
  }
  return f;
}

// ---------------- WMMA GEMM:  C = A * W^T, wave computes a 16 x (16*NT) strip ----------------
// Software-pipelined K loop: fragments for step k+1 are loaded while WMMAs consume step k.
// EPI 0: plain f32 store | EPI 1: softplus(v + bias[n]) | EPI 2: residual + transposed (B,DM,L)
// EPI 3: f32 store + f16 copy of the dt columns (n < cDTR) for the delta GEMM
template <int KMAX, int NT, int EPI>
__global__ void wmma_gemm_kernel(const _Float16* __restrict__ A, int lda,
                                 const _Float16* __restrict__ W, int ldw,
                                 float* __restrict__ Cp, int M, int Ncols,
                                 const float* __restrict__ bias,
                                 const float* __restrict__ resid,
                                 float* __restrict__ outp,
                                 _Float16* __restrict__ dt16) {
  const int ntN    = Ncols / (16 * NT);
  const int ntiles = (M >> 4) * ntN;
  const int tile   = blockIdx.x * (blockDim.x >> 5) + (threadIdx.x >> 5);
  if (tile >= ntiles) return;                 // uniform per-wave exit: EXEC all-1s for WMMA
  const int lane = threadIdx.x & 31;
  const int half = lane >> 4;
  const int jn   = lane & 15;
  const int m0 = (tile / ntN) << 4;
  const int n0 = (tile % ntN) * 16 * NT;

  const _Float16* arow = A + (long)(m0 + jn) * lda;
  const _Float16* brow[NT];
#pragma unroll
  for (int s = 0; s < NT; ++s) brow[s] = W + (long)(n0 + s * 16 + jn) * ldw;

  v8f acc[NT];
#pragma unroll
  for (int s = 0; s < NT; ++s) acc[s] = (v8f){};

  constexpr int KMAIN = (KMAX / 32) * 32;
  if constexpr (KMAIN > 0) {
    // prologue: load fragments for k0 = 0
    v16h af = frag_a16(arow, 0, half);
    v16h bf[NT];
#pragma unroll
    for (int s = 0; s < NT; ++s) bf[s] = frag_b16(brow[s], 0, half);

    for (int k0 = 0; k0 < KMAIN; k0 += 32) {
      v16h afn;
      v16h bfn[NT];
      const int kn = k0 + 32;
      if (kn < KMAIN) {                       // prefetch next step while WMMAs run
        afn = frag_a16(arow, kn, half);
#pragma unroll
        for (int s = 0; s < NT; ++s) bfn[s] = frag_b16(brow[s], kn, half);
      }
#pragma unroll
      for (int s = 0; s < NT; ++s)
        acc[s] = __builtin_amdgcn_wmma_f32_16x16x32_f16(false, af, false, bf[s],
                                                        (short)0, acc[s], false, false);
      af = afn;
#pragma unroll
      for (int s = 0; s < NT; ++s) bf[s] = bfn[s];
    }
  }
  if constexpr (KMAX % 32 != 0) {
    v16h bf[NT];
#pragma unroll
    for (int s = 0; s < NT; ++s) bf[s] = frag_b16_tail<KMAX>(brow[s], half);
    v16h af = frag_a16_tail<KMAX>(arow, half);
#pragma unroll
    for (int s = 0; s < NT; ++s)
      acc[s] = __builtin_amdgcn_wmma_f32_16x16x32_f16(false, af, false, bf[s],
                                                      (short)0, acc[s], false, false);
  }

  // C/D layout: lane<16 -> (M=m0+i, N=n0+lane); lane>=16 -> (M=m0+8+i, N=n0+lane-16)
#pragma unroll
  for (int s = 0; s < NT; ++s) {
    const int nn = n0 + s * 16 + jn;
#pragma unroll
    for (int i = 0; i < 8; ++i) {
      const int m = m0 + half * 8 + i;
      float v = acc[s][i];
      if (EPI == 1) {
        float t = v + bias[nn];
        Cp[(long)m * Ncols + nn] = (t > 20.f) ? t : log1pf(__expf(t));
      } else if (EPI == 2) {
        const int b = m / cL, l = m - b * cL;
        const long o = (long)b * cDM * cL + (long)nn * cL + l;
        outp[o] = v + resid[o];
      } else {
        Cp[(long)m * Ncols + nn] = v;
        if (EPI == 3 && s * 16 < cDTR)          // dt columns (subtile 0) also stored f16
          dt16[(long)m * cDTR + nn] = (_Float16)v;
      }
    }
  }
}

// ---------------- LDS-tiled transpose + f16 convert: x (B,DM,L) -> h16 (B*L, DM) ----------------
__global__ void transpose_x_kernel(const float* __restrict__ x, _Float16* __restrict__ h) {
  __shared__ float tile[32][33];                   // +1 pad: no LDS bank conflicts
  const int bL  = blockIdx.x;
  const int bDM = blockIdx.y;
  const int b   = blockIdx.z;
  const int tx = threadIdx.x & 31, ty = threadIdx.x >> 5;   // 256 threads: ty = 0..7
  const int l0 = bL * 32, dm0 = bDM * 32;
#pragma unroll
  for (int i = 0; i < 4; ++i) {
    const int dm = dm0 + ty + i * 8;
    tile[ty + i * 8][tx] = x[(long)b * cDM * cL + (long)dm * cL + (l0 + tx)];
  }
  __syncthreads();
#pragma unroll
  for (int i = 0; i < 4; ++i) {
    const int l = l0 + ty + i * 8;
    h[((long)b * cL + l) * cDM + (dm0 + tx)] = (_Float16)tile[tx][ty + i * 8];
  }
}

// ---------------- vectorized f32 -> f16 convert (n % 4 == 0) ----------------
__global__ void f32_to_f16_kernel(const float* __restrict__ src, _Float16* __restrict__ dst, long n4) {
  const long i = (long)blockIdx.x * blockDim.x + threadIdx.x;
  if (i >= n4) return;
  const float4 v = *(const float4*)(src + i * 4);
  _Float16* d = dst + i * 4;
  d[0] = (_Float16)v.x; d[1] = (_Float16)v.y; d[2] = (_Float16)v.z; d[3] = (_Float16)v.w;
}

// ---------------- causal depthwise conv + SiLU -> f16 u (branch-local, optional flip) ----------------
__global__ void conv_silu_kernel(const float* __restrict__ xz,
                                 const float* __restrict__ conv_w,
                                 const float* __restrict__ conv_b,
                                 _Float16* __restrict__ u, int flip) {
  const long idx = (long)blockIdx.x * blockDim.x + threadIdx.x;
  if (idx >= (long)cB * cL * cDIN) return;
  const int d = (int)(idx % cDIN);
  const int l = (int)((idx / cDIN) % cL);
  const int b = (int)(idx / ((long)cDIN * cL));

  float acc = conv_b[d];
#pragma unroll
  for (int k = 0; k < cK; ++k) {
    const int t = l - (cK - 1) + k;
    if (t >= 0) {
      const int lsrc = flip ? (cL - 1 - t) : t;
      acc = fmaf(conv_w[d * cK + k], xz[((long)b * cL + lsrc) * cXZC + d], acc);
    }
  }
  const float s = acc / (1.f + __expf(-acc));      // SiLU
  u[((long)b * cL + l) * cDIN + d] = (_Float16)s;
}

// ---------------- selective scan: 16 lanes (one per state n) per (b,d) series ----------------
__global__ void ssm_scan_kernel(const _Float16* __restrict__ u,
                                const float* __restrict__ delta,
                                const float* __restrict__ xdbl,
                                const float* __restrict__ xz,     // z gate: cols 512..1023
                                const float* __restrict__ A_log,
                                const float* __restrict__ Dp,
                                float* __restrict__ y, int flip) {
  const int dchunks = cDIN / 16;                  // 32
  const int b  = blockIdx.x / dchunks;
  const int dc = blockIdx.x % dchunks;
  const int d  = dc * 16 + (threadIdx.x >> 4);
  const int n  = threadIdx.x & 15;

  const float Adn = -__expf(A_log[d * cN + n]);   // A = -exp(A_log)
  const float Dd  = Dp[d];
  float h = 0.f;

  for (int t = 0; t < cL; ++t) {
    const long row = (long)b * cL + t;
    if (t + 16 < cL) {                            // gfx1250 prefetch path
      const long prow = row + 16;
      __builtin_prefetch(&delta[prow * cDIN + d], 0, 0);
      __builtin_prefetch(&xdbl[prow * cXDC + cDTR], 0, 0);
    }
    const float dt = delta[row * cDIN + d];       // broadcast across the 16-lane group
    const float ut = (float)u[row * cDIN + d];
    const float Bn = xdbl[row * cXDC + cDTR + n];
    const float Cn = xdbl[row * cXDC + cDTR + cN + n];

    const float dA = __expf(dt * Adn);
    h = fmaf(dA, h, dt * ut * Bn);

    float yp = h * Cn;                            // contract over n within 16-lane group
    yp += __shfl_xor(yp, 1);
    yp += __shfl_xor(yp, 2);
    yp += __shfl_xor(yp, 4);
    yp += __shfl_xor(yp, 8);

    if (n == 0) {
      const int lout = flip ? (cL - 1 - t) : t;   // un-flip backward branch on output
      const long orow = (long)b * cL + lout;
      const float zt = xz[orow * cXZC + cDIN + d];
      const float sz = zt / (1.f + __expf(-zt));
      const float val = (yp + ut * Dd) * sz;
      if (flip) y[orow * cDIN + d] += val;        // accumulate onto forward result
      else      y[orow * cDIN + d]  = val;
    }
  }
}

// ---------------- launch ----------------
extern "C" void kernel_launch(void* const* d_in, const int* in_sizes, int n_in,
                              void* d_out, int out_size, void* d_ws, size_t ws_size,
                              hipStream_t stream) {
  const float* x         = (const float*)d_in[0];   // (B, DM, 1, L)
  const float* in_proj_w = (const float*)d_in[1];   // (1024, 256)
  const float* conv_w_f  = (const float*)d_in[2];
  const float* conv_b_f  = (const float*)d_in[3];
  const float* xproj_w_f = (const float*)d_in[4];   // (48, 512)
  const float* dt_w_f    = (const float*)d_in[5];   // (512, 16)
  const float* dt_b_f    = (const float*)d_in[6];
  const float* A_log_f   = (const float*)d_in[7];
  const float* D_f       = (const float*)d_in[8];
  const float* conv_w_b  = (const float*)d_in[9];
  const float* conv_b_b  = (const float*)d_in[10];
  const float* xproj_w_b = (const float*)d_in[11];
  const float* dt_w_b    = (const float*)d_in[12];
  const float* dt_b_b    = (const float*)d_in[13];
  const float* A_log_b   = (const float*)d_in[14];
  const float* D_b       = (const float*)d_in[15];
  const float* out_proj_w= (const float*)d_in[16];  // (256, 512)
  float* out = (float*)d_out;

  // ---------------- workspace carve-up (all chunk sizes are multiples of 256 B) ----------------
  char* p = (char*)d_ws;
  auto take = [&](size_t bytes) { char* q = p; p += bytes; return q; };
  _Float16* hT16    = (_Float16*)take((size_t)cROWS * cDM  * 2);   // x transposed, f16
  float*    xz      = (float*)   take((size_t)cROWS * cXZC * 4);
  _Float16* u16_f   = (_Float16*)take((size_t)cROWS * cDIN * 2);
  _Float16* u16_b   = (_Float16*)take((size_t)cROWS * cDIN * 2);
  float*    xdbl_f  = (float*)   take((size_t)cROWS * cXDC * 4);
  float*    xdbl_b  = (float*)   take((size_t)cROWS * cXDC * 4);
  _Float16* dt16_f  = (_Float16*)take((size_t)cROWS * cDTR * 2);
  _Float16* dt16_b  = (_Float16*)take((size_t)cROWS * cDTR * 2);
  float*    delta_f = (float*)   take((size_t)cROWS * cDIN * 4);
  float*    delta_b = (float*)   take((size_t)cROWS * cDIN * 4);
  float*    ybuf    = (float*)   take((size_t)cROWS * cDIN * 4);
  _Float16* y16     = (_Float16*)take((size_t)cROWS * cDIN * 2);
  _Float16* w16_in  = (_Float16*)take((size_t)cXZC * cDM  * 2);
  _Float16* w16_xp_f= (_Float16*)take((size_t)cXDC * cDIN * 2);
  _Float16* w16_xp_b= (_Float16*)take((size_t)cXDC * cDIN * 2);
  _Float16* w16_dt_f= (_Float16*)take((size_t)cDIN * cDTR * 2);
  _Float16* w16_dt_b= (_Float16*)take((size_t)cDIN * cDTR * 2);
  _Float16* w16_out = (_Float16*)take((size_t)cDM  * cDIN * 2);

  const dim3 blk(256);
  auto gblocks = [](long tiles) { return dim3((unsigned)((tiles + 7) / 8)); };
  auto cvt = [&](const float* s, _Float16* d, long n) {
    const long n4 = n / 4;
    f32_to_f16_kernel<<<dim3((unsigned)((n4 + 255) / 256)), blk, 0, stream>>>(s, d, n4);
  };

  // 0) one-time f16 conversions: weights + transposed activations
  cvt(in_proj_w,  w16_in,   (long)cXZC * cDM);
  cvt(xproj_w_f,  w16_xp_f, (long)cXDC * cDIN);
  cvt(xproj_w_b,  w16_xp_b, (long)cXDC * cDIN);
  cvt(dt_w_f,     w16_dt_f, (long)cDIN * cDTR);
  cvt(dt_w_b,     w16_dt_b, (long)cDIN * cDTR);
  cvt(out_proj_w, w16_out,  (long)cDM * cDIN);
  transpose_x_kernel<<<dim3(cL / 32, cDM / 32, cB), blk, 0, stream>>>(x, hT16);

  // 1) xz = hT @ in_proj_w^T           M=8192 N=1024 K=256, 16x64 strips
  wmma_gemm_kernel<cDM, 4, 0><<<gblocks((cROWS / 16) * (cXZC / 64)), blk, 0, stream>>>(
      hT16, cDM, w16_in, cDM, xz, cROWS, cXZC, nullptr, nullptr, nullptr, nullptr);

  // 2) depthwise causal conv + SiLU -> f16 u (backward branch in flipped-time coords)
  {
    const long total = (long)cB * cL * cDIN;
    const dim3 g((unsigned)((total + 255) / 256));
    conv_silu_kernel<<<g, blk, 0, stream>>>(xz, conv_w_f, conv_b_f, u16_f, 0);
    conv_silu_kernel<<<g, blk, 0, stream>>>(xz, conv_w_b, conv_b_b, u16_b, 1);
  }

  // 3) x_dbl = u @ xproj_w^T           M=8192 N=48 K=512; dt cols also stored f16 (EPI 3)
  wmma_gemm_kernel<cDIN, 3, 3><<<gblocks((cROWS / 16) * (cXDC / 48)), blk, 0, stream>>>(
      u16_f, cDIN, w16_xp_f, cDIN, xdbl_f, cROWS, cXDC, nullptr, nullptr, nullptr, dt16_f);
  wmma_gemm_kernel<cDIN, 3, 3><<<gblocks((cROWS / 16) * (cXDC / 48)), blk, 0, stream>>>(
      u16_b, cDIN, w16_xp_b, cDIN, xdbl_b, cROWS, cXDC, nullptr, nullptr, nullptr, dt16_b);

  // 4) delta = softplus(dt @ dt_w^T + dt_b)   M=8192 N=512 K=16 (compile-time padded tail)
  wmma_gemm_kernel<cDTR, 4, 1><<<gblocks((cROWS / 16) * (cDIN / 64)), blk, 0, stream>>>(
      dt16_f, cDTR, w16_dt_f, cDTR, delta_f, cROWS, cDIN, dt_b_f, nullptr, nullptr, nullptr);
  wmma_gemm_kernel<cDTR, 4, 1><<<gblocks((cROWS / 16) * (cDIN / 64)), blk, 0, stream>>>(
      dt16_b, cDTR, w16_dt_b, cDTR, delta_b, cROWS, cDIN, dt_b_b, nullptr, nullptr, nullptr);

  // 5) selective scans: forward writes ybuf, backward accumulates (un-flipped)
  {
    const dim3 g((unsigned)(cB * (cDIN / 16)));   // 128 blocks x 256 threads
    ssm_scan_kernel<<<g, blk, 0, stream>>>(u16_f, delta_f, xdbl_f, xz, A_log_f, D_f, ybuf, 0);
    ssm_scan_kernel<<<g, blk, 0, stream>>>(u16_b, delta_b, xdbl_b, xz, A_log_b, D_b, ybuf, 1);
  }

  // 6) y -> f16, then out = y @ out_proj_w^T + x_res, stored transposed to (B, DM, L)
  cvt(ybuf, y16, (long)cROWS * cDIN);
  wmma_gemm_kernel<cDIN, 4, 2><<<gblocks((cROWS / 16) * (cDM / 64)), blk, 0, stream>>>(
      y16, cDIN, w16_out, cDIN, nullptr, cROWS, cDM, nullptr, x, out, nullptr);
}
